// Skiptensor_21998822490681
// MI455X (gfx1250) — matrix operand order repaired and checked by reading, not compiled, not run on hardware
//
#include <hip/hip_runtime.h>

#define B_C     4
#define S_C     4096
#define DIMS_C  1024
#define HEAD_C  16
#define HD_C    64
#define SKIP_C  4
#define SK_C    (S_C / SKIP_C)   // 1024 keys per (b,h)

typedef __bf16 bf16_t;
typedef __attribute__((ext_vector_type(16))) __bf16 v16bf;
typedef __attribute__((ext_vector_type(8)))  __bf16 v8bf;
typedef __attribute__((ext_vector_type(8)))  float  v8f;
typedef __attribute__((ext_vector_type(4)))  float  v4f;

// Fragment unions: let b128 loads land directly in fragment register quads
// (no v_mov_b16 sub-register shuffling).
union V16 { v16bf v; v8bf h[2]; unsigned int u[8]; };
union V8  { v8bf  v; unsigned int u[4]; };

// float -> bf16 RNE as integer ops; pack two at a time into a dword.
__device__ __forceinline__ unsigned int bf_bits(float f) {
  unsigned int u = __builtin_bit_cast(unsigned int, f);
  u += 0x7FFFu + ((u >> 16) & 1u);
  return u >> 16;
}
__device__ __forceinline__ unsigned int pack_bf2(float lo, float hi) {
  return bf_bits(lo) | (bf_bits(hi) << 16);
}
__device__ __forceinline__ bf16_t f2bf(float f) {
  unsigned short s = (unsigned short)bf_bits(f);
  return __builtin_bit_cast(bf16_t, s);
}

__device__ __forceinline__ v8f wmma_bf16(v16bf a, v16bf b, v8f c) {
  // 8 args: (neg_a, A, neg_b, B, c_mod, C, reuse_a, reuse_b)
  return __builtin_amdgcn_wmma_f32_16x16x32_bf16(false, a, false, b, (short)0, c,
                                                 false, false);
}

// -----------------------------------------------------------------------------
// Streaming f32 -> bf16 converter (8 elems/thread). n must be a multiple of
// 2048 (true for all buffers here). Memory-bound by design.
// -----------------------------------------------------------------------------
__global__ __launch_bounds__(256) void f32_to_bf16_kernel(
    const float* __restrict__ src, bf16_t* __restrict__ dst)
{
  const size_t i = ((size_t)blockIdx.x * 256 + threadIdx.x) * 8;
  v4f f0 = *(const v4f*)(src + i);
  v4f f1 = *(const v4f*)(src + i + 4);
  V8 o;
  o.u[0] = pack_bf2(f0[0], f0[1]);
  o.u[1] = pack_bf2(f0[2], f0[3]);
  o.u[2] = pack_bf2(f1[0], f1[1]);
  o.u[3] = pack_bf2(f1[2], f1[3]);
  *(v8bf*)(dst + i) = o.v;
}

// -----------------------------------------------------------------------------
// Y[i, e] = sum_d X[i*rowStride, d] * W[e, d] + bias[e]   (einsum 'bsd,ed')
// X and W pre-converted to bf16: inner loop is pure b128 loads + WMMA.
// Block = 128 threads (4 waves); each wave owns 32(M) x 64(N) so both
// A-fragments reuse every B-fragment (8 WMMA per 6 vector loads).
// grid.x = M/32, grid.y = DIMS/256.
// OUT_MODE: 0 = f32 row-major, 1 = bf16 row-major, 2 = bf16 transposed V^T
//           laid out as (b, e, key) with key-length SK_C.
// -----------------------------------------------------------------------------
template <int OUT_MODE>
__global__ __launch_bounds__(128) void gemm_bf16_wmma(
    const bf16_t* __restrict__ X, int rowStride,
    const bf16_t* __restrict__ W, const float* __restrict__ bias,
    void* __restrict__ Yraw)
{
  const int lane = threadIdx.x & 31;
  const int wave = threadIdx.x >> 5;
  const int half = lane >> 4;     // K split across lane halves
  const int lr   = lane & 15;     // A: row / B,D: column

  const int rowBase = blockIdx.x * 32;
  const int colBase = blockIdx.y * 256 + wave * 64;

  const bf16_t* xr0 = X + (size_t)(rowBase + lr)      * rowStride * DIMS_C;
  const bf16_t* xr1 = X + (size_t)(rowBase + 16 + lr) * rowStride * DIMS_C;

  v8f acc[2][4] = {};

  for (int k0 = 0; k0 < DIMS_C; k0 += 32) {
    __builtin_prefetch(xr0 + k0 + 128, 0, 1);
    __builtin_prefetch(xr1 + k0 + 128, 0, 1);
    // A fragments: lane lr = row; elem j -> K=k0+half*16+j
    V16 a0, a1;
    a0.h[0] = *(const v8bf*)(xr0 + k0 + half * 16);
    a0.h[1] = *(const v8bf*)(xr0 + k0 + half * 16 + 8);
    a1.h[0] = *(const v8bf*)(xr1 + k0 + half * 16);
    a1.h[1] = *(const v8bf*)(xr1 + k0 + half * 16 + 8);
#pragma unroll
    for (int nt = 0; nt < 4; ++nt) {
      // B fragment: lane lr = output col e; elem j -> K = k0 + half*16 + j
      const bf16_t* wp =
          W + (size_t)(colBase + nt * 16 + lr) * DIMS_C + k0 + half * 16;
      v16bf b = *(const v16bf*)wp;
      acc[0][nt] = wmma_bf16(a0.v, b, acc[0][nt]);
      acc[1][nt] = wmma_bf16(a1.v, b, acc[1][nt]);
    }
  }

  // D layout: lane -> col lr; VGPR r -> row (mt*16 + half*8 + r)
#pragma unroll
  for (int mt = 0; mt < 2; ++mt) {
#pragma unroll
    for (int nt = 0; nt < 4; ++nt) {
      const int col = colBase + nt * 16 + lr;
      const float bv = bias[col];
      if (OUT_MODE == 2) {
        // V^T store: this lane holds 8 consecutive keys of one dim -> one
        // contiguous 16B store at VT[(b*DIMS + e)*SK + key0].
        const int row0 = rowBase + mt * 16 + half * 8;
        const int bb   = row0 / SK_C;
        const int key0 = row0 - bb * SK_C;
        V8 s;
#pragma unroll
        for (int r = 0; r < 4; ++r)
          s.u[r] = pack_bf2(acc[mt][nt][2 * r] + bv, acc[mt][nt][2 * r + 1] + bv);
        *(v8bf*)((bf16_t*)Yraw + ((size_t)bb * DIMS_C + col) * SK_C + key0) = s.v;
      } else {
#pragma unroll
        for (int r = 0; r < 8; ++r) {
          const int row = rowBase + mt * 16 + half * 8 + r;
          const float vv = acc[mt][nt][r] + bv;
          if (OUT_MODE == 1)
            ((bf16_t*)Yraw)[(size_t)row * DIMS_C + col] = f2bf(vv);
          else
            ((float*)Yraw)[(size_t)row * DIMS_C + col] = vv;
        }
      }
    }
  }
}

// -----------------------------------------------------------------------------
// Flash-style attention over 1024 subsampled keys. Per wave: 16 queries of one
// (b, h). S^T = K_blk @ Q^T so the score D-layout (lane=query, VGPRs=keys) IS
// the A-layout for P @ V. V is pre-transposed so each B-fragment is two
// aligned v8bf loads. grid = (S/64, HEAD, B), block = 128 (4 waves).
// -----------------------------------------------------------------------------
__global__ __launch_bounds__(128) void attn_wmma_kernel(
    const bf16_t* __restrict__ Q, const bf16_t* __restrict__ K,
    const bf16_t* __restrict__ VT, bf16_t* __restrict__ O, float scale)
{
  const int lane = threadIdx.x & 31;
  const int wave = threadIdx.x >> 5;
  const int half = lane >> 4;
  const int lr   = lane & 15;

  const int b = blockIdx.z, h = blockIdx.y;
  const int qbase = (blockIdx.x * 4 + wave) * 16;

  const bf16_t* Qp  = Q  + ((size_t)b * S_C + qbase) * DIMS_C + h * HD_C;
  const bf16_t* Kp  = K  + (size_t)b * SK_C * DIMS_C + h * HD_C;
  const bf16_t* VTp = VT + ((size_t)b * DIMS_C + h * HD_C) * SK_C;

  // Q^T as two B fragments (head-dim chunks). elem j -> K = c*32 + half*16 + j
  v16bf qf[2];
#pragma unroll
  for (int c = 0; c < 2; ++c)
    qf[c] = *(const v16bf*)(Qp + (size_t)lr * DIMS_C + c * 32 + half * 16);

  v8f   o[4] = {};
  float m_run = -3.0e38f;
  float l_run = 0.0f;

  for (int kb = 0; kb < SK_C; kb += 32) {
    // ---- S^T tiles: st[mt] = (keys kb+mt*16..+15) x (16 queries) ----
    v8f st[2] = {};
#pragma unroll
    for (int c = 0; c < 2; ++c) {
#pragma unroll
      for (int mt = 0; mt < 2; ++mt) {
        const bf16_t* kp = Kp + (size_t)(kb + mt * 16 + lr) * DIMS_C + c * 32;
        V16 ka;
        ka.h[0] = *(const v8bf*)(kp + half * 16);      // K = half*16 + 0..7
        ka.h[1] = *(const v8bf*)(kp + half * 16 + 8);  // K = half*16 + 8..15
        st[mt] = wmma_bf16(ka.v, qf[c], st[mt]);
      }
    }
#pragma unroll
    for (int mt = 0; mt < 2; ++mt)
#pragma unroll
      for (int r = 0; r < 8; ++r) st[mt][r] *= scale;

    // ---- online softmax; this lane's query = lr, partner = lane^16 ----
    float mloc = -3.0e38f;
#pragma unroll
    for (int mt = 0; mt < 2; ++mt)
#pragma unroll
      for (int r = 0; r < 8; ++r) mloc = fmaxf(mloc, st[mt][r]);
    mloc = fmaxf(mloc, __shfl_xor(mloc, 16, 32));

    const float mnew = fmaxf(m_run, mloc);
    const float corr = __expf(m_run - mnew);

    v8f p0, p1;
    float psum = 0.0f;
#pragma unroll
    for (int r = 0; r < 8; ++r) {
      p0[r] = __expf(st[0][r] - mnew); psum += p0[r];
      p1[r] = __expf(st[1][r] - mnew); psum += p1[r];
    }
    psum += __shfl_xor(psum, 16, 32);
    l_run = l_run * corr + psum;
    m_run = mnew;

    // Rescale O accumulators: O rows are queries half*8+r; query q's corr
    // lives in lanes q / q+16 -> gather via shuffle (ds_bpermute).
#pragma unroll
    for (int r = 0; r < 8; ++r) {
      float cr = __shfl(corr, (half << 3) + r, 32);
#pragma unroll
      for (int nt = 0; nt < 4; ++nt) o[nt][r] *= cr;
    }

    // Pack P (16 queries x 32 keys) straight into A layout via dword packs.
    V16 pa;
#pragma unroll
    for (int i = 0; i < 4; ++i) {
      pa.u[i]     = pack_bf2(p0[2 * i], p0[2 * i + 1]);  // keys half*8 + 0..7
      pa.u[4 + i] = pack_bf2(p1[2 * i], p1[2 * i + 1]);  // keys 16+half*8+0..7
    }

    // ---- O += P @ V_blk : B-fragment = two v8bf of V^T (keys half*8, 16+half*8) ----
#pragma unroll
    for (int nt = 0; nt < 4; ++nt) {
      const bf16_t* vp = VTp + (size_t)(nt * 16 + lr) * SK_C + kb;
      V16 vb;
      vb.h[0] = *(const v8bf*)(vp + half * 8);
      vb.h[1] = *(const v8bf*)(vp + 16 + half * 8);
      o[nt] = wmma_bf16(pa.v, vb.v, o[nt]);
    }
  }

  // ---- normalize by row sums and store bf16 ----
#pragma unroll
  for (int r = 0; r < 8; ++r) {
    float inv = 1.0f / __shfl(l_run, (half << 3) + r, 32);
    const size_t row = (size_t)b * S_C + qbase + half * 8 + r;
#pragma unroll
    for (int nt = 0; nt < 4; ++nt)
      O[row * DIMS_C + h * HD_C + nt * 16 + lr] = f2bf(o[nt][r] * inv);
  }
}

// -----------------------------------------------------------------------------
extern "C" void kernel_launch(void* const* d_in, const int* in_sizes, int n_in,
                              void* d_out, int out_size, void* d_ws, size_t ws_size,
                              hipStream_t stream)
{
  (void)in_sizes; (void)n_in; (void)out_size; (void)ws_size;

  const float* x  = (const float*)d_in[0];
  const float* Wq = (const float*)d_in[1];
  const float* bq = (const float*)d_in[2];
  const float* Wk = (const float*)d_in[3];
  const float* bk = (const float*)d_in[4];
  const float* Wv = (const float*)d_in[5];
  const float* bv = (const float*)d_in[6];
  const float* Wo = (const float*)d_in[7];
  const float* bo = (const float*)d_in[8];
  // d_in[9] = skip_pattern (== 4 per setup; baked into SKIP_C)

  // Workspace (bf16): xbf 32MB | 4x W 2MB | Q 32MB | K 8MB | V^T 8MB | AO 32MB
  const size_t N_X = (size_t)B_C * S_C * DIMS_C;     // 16.7M
  const size_t N_W = (size_t)DIMS_C * DIMS_C;        // 1.05M
  const size_t N_K = (size_t)B_C * SK_C * DIMS_C;    // 4.2M

  bf16_t* xb  = (bf16_t*)d_ws;
  bf16_t* wqb = xb  + N_X;
  bf16_t* wkb = wqb + N_W;
  bf16_t* wvb = wkb + N_W;
  bf16_t* wob = wvb + N_W;
  bf16_t* Qb  = wob + N_W;
  bf16_t* Kb  = Qb  + N_X;
  bf16_t* VTb = Kb  + N_K;
  bf16_t* AO  = VTb + N_K;

  const dim3 blk128(128);
  const float scale = 0.03125f;  // DIMS^-0.5 = 1/32

  // One-shot bf16 conversion of activations + weights (memory-bound stream).
  f32_to_bf16_kernel<<<N_X / 2048, 256, 0, stream>>>(x,  xb);
  f32_to_bf16_kernel<<<N_W / 2048, 256, 0, stream>>>(Wq, wqb);
  f32_to_bf16_kernel<<<N_W / 2048, 256, 0, stream>>>(Wk, wkb);
  f32_to_bf16_kernel<<<N_W / 2048, 256, 0, stream>>>(Wv, wvb);
  f32_to_bf16_kernel<<<N_W / 2048, 256, 0, stream>>>(Wo, wob);

  // Projections (K/V fold the ::4 subsampling via rowStride=4).
  gemm_bf16_wmma<1><<<dim3((B_C * S_C) / 32, DIMS_C / 256), blk128, 0, stream>>>(
      xb, 1, wqb, bq, Qb);
  gemm_bf16_wmma<1><<<dim3((B_C * SK_C) / 32, DIMS_C / 256), blk128, 0, stream>>>(
      xb, SKIP_C, wkb, bk, Kb);
  gemm_bf16_wmma<2><<<dim3((B_C * SK_C) / 32, DIMS_C / 256), blk128, 0, stream>>>(
      xb, SKIP_C, wvb, bv, VTb);

  // Flash attention (WMMA for K@Q^T and P@V).
  attn_wmma_kernel<<<dim3(S_C / 64, HEAD_C, B_C), blk128, 0, stream>>>(
      Qb, Kb, VTb, AO, scale);

  // Output projection -> f32 d_out.
  gemm_bf16_wmma<0><<<dim3((B_C * S_C) / 32, DIMS_C / 256), blk128, 0, stream>>>(
      AO, 1, wob, bo, d_out);
}
